// RecursiveAttentiveEncoder_24068996726963
// MI455X (gfx1250) — compile-verified, matrix operand order checked by reading, not run on hardware
//
#include <hip/hip_runtime.h>
#include <hip/hip_bf16.h>

// ---------------------------------------------------------------------------
// Problem constants (from reference)
#define B_    1024
#define T_    34
#define EDIM  128

// LDS row strides (bf16 elems), padded +8 (16B) so tile fetches stripe banks.
#define SA   2056   // 16 x 2048 leaf/level buffer
#define SB   1032   // 16 x 1024 level buffer
#define SH1  264    // 16 x 256 hidden (tree MLP)
#define SX   264    // 16 x 256 x (stmt MLP input)
#define SH2  520    // 16 x 512 hidden (stmt MLP)

typedef __attribute__((ext_vector_type(16))) __bf16 v16bf;
typedef __attribute__((ext_vector_type(8)))  float  v8f;
typedef __attribute__((ext_vector_type(4)))  int    v4i;

union TileBF { v16bf v; uint4 q[2]; };

// f32 -> bf16 bits (round-to-nearest-even via HIP's native conversion)
__device__ inline unsigned short f2us(float f) {
  __hip_bfloat16 b = __float2bfloat16(f);
  return __builtin_bit_cast(unsigned short, b);
}

// Async global->LDS copy path (CDNA5): builtin takes (global int4*, lds int4*,
// imm offset, imm cpol) per the compiler's signature. Guarded so the file
// still compiles on toolchains without the builtin.
#if __has_builtin(__builtin_amdgcn_global_load_async_to_lds_b128) && \
    __has_builtin(__builtin_amdgcn_s_wait_asynccnt)
#define USE_ASYNC_LDS 1
typedef __attribute__((address_space(1))) v4i* gv4i_p;
typedef __attribute__((address_space(3))) v4i* lv4i_p;
#else
#define USE_ASYNC_LDS 0
#endif

// Load one 16x32 bf16 WMMA operand tile (A layout; B uses identical layout
// when weights are stored transposed [out][in]).
//  lane L (<16): row M=L, K-chunks {k0..k0+7} and {k0+16..k0+23}
//  lane 16+L   : row M=L, K-chunks {k0+8..k0+15} and {k0+24..k0+31}
__device__ inline TileBF load_tile(const unsigned short* base, int ld) {
  const int lane = threadIdx.x & 31;
  const int r    = lane & 15;
  const int kh   = (lane & 16) >> 1;   // 0 or 8
  TileBF t;
  t.q[0] = *reinterpret_cast<const uint4*>(base + r * ld + kh);
  t.q[1] = *reinterpret_cast<const uint4*>(base + r * ld + kh + 16);
  return t;
}

__device__ inline v8f wmma_bf16(const TileBF& a, const TileBF& b, v8f c) {
  return __builtin_amdgcn_wmma_f32_16x16x32_bf16(
      /*neg_a=*/false, a.v, /*neg_b=*/false, b.v,
      /*c_mod=*/(short)0, c, /*reuse_a=*/false, /*reuse_b=*/false);
}

// ---------------------------------------------------------------------------
// K0a: W[e][k][n] (f32) -> Wt[e][n][k] (bf16)  (transpose so B-tiles load like A)
__global__ __launch_bounds__(256) void wcvt_kernel(
    const float* __restrict__ W, unsigned short* __restrict__ Wt,
    int KN, int K, int N, int total) {
  int i = blockIdx.x * 256 + threadIdx.x;
  if (i >= total) return;
  int e   = i / KN;
  int rem = i - e * KN;
  int k   = rem / N;
  int n   = rem - k * N;
  Wt[(e * N + n) * K + k] = f2us(W[i]);
}

// K0b: plain f32 -> bf16 convert (embedding table, enables async LDS gather)
__global__ __launch_bounds__(256) void ecvt_kernel(
    const float* __restrict__ E, unsigned short* __restrict__ O, int total) {
  int i = blockIdx.x * 256 + threadIdx.x;
  if (i < total) O[i] = f2us(E[i]);
}

// ---------------------------------------------------------------------------
// K1: tree reduction. grid = (64 batch-blocks, 34 trees), 256 threads (8 waves)
__global__ __launch_bounds__(256) void tree_kernel(
    const int* __restrict__ leaf_idx, const int* __restrict__ nf_ids,
    const float* __restrict__ ent_emb, const unsigned short* __restrict__ ent16,
    const unsigned short* __restrict__ W1t, const float* __restrict__ b1,
    const unsigned short* __restrict__ W2t, const float* __restrict__ b2,
    float* __restrict__ roots) {
  __shared__ unsigned short bufA[16 * SA];
  __shared__ unsigned short bufB[16 * SB];
  __shared__ unsigned short h1b[16 * SH1];
  __shared__ int ids_s[16 * 15];

  const int bb   = blockIdx.x;
  const int t    = blockIdx.y;
  const int tid  = threadIdx.x;
  const int wave = tid >> 5;
  const int lane = tid & 31;
  const int col  = lane & 15;
  const int rb   = (lane & 16) >> 1;   // 0 or 8 (C/D row base)

  if (tid < 240) {
    int r = tid / 15, k = tid - r * 15;
    ids_s[tid] = nf_ids[((bb * 16 + r) * T_ + t) * 15 + k];
  }
  { // gather leaf embeddings: thread = (row r, leaf l)
    int r = tid >> 4, l = tid & 15;
    int idx = leaf_idx[((bb * 16 + r) * T_ + t) * 16 + l];
#if USE_ASYNC_LDS
    const unsigned short* src = ent16 + idx * EDIM;
    unsigned short* dst = bufA + r * SA + l * EDIM;
    #pragma unroll
    for (int ch = 0; ch < 16; ++ch)   // 16 x 16B = one 256B embedding row
      __builtin_amdgcn_global_load_async_to_lds_b128(
          (gv4i_p)(src + ch * 8), (lv4i_p)(dst + ch * 8), 0, 0);
    __builtin_amdgcn_s_wait_asynccnt(0);
#else
    const float* src = ent_emb + idx * EDIM;
    unsigned short* dst = bufA + r * SA + l * EDIM;
    #pragma unroll
    for (int d = 0; d < EDIM; d += 4) {
      float4 v = *reinterpret_cast<const float4*>(src + d);
      dst[d + 0] = f2us(v.x); dst[d + 1] = f2us(v.y);
      dst[d + 2] = f2us(v.z); dst[d + 3] = f2us(v.w);
    }
#endif
  }
  __syncthreads();

  int off = 0;
  for (int lvl = 0; lvl < 4; ++lvl) {
    const int n = 8 >> lvl;
    unsigned short* bin  = (lvl & 1) ? bufB : bufA;
    unsigned short* bout = (lvl & 1) ? bufA : bufB;
    const int sin  = (lvl & 1) ? SB : SA;
    const int sout = (lvl & 1) ? SA : SB;
    for (int j = 0; j < n; ++j) {
      int idv[8];
      #pragma unroll
      for (int g = 0; g < 8; ++g) idv[g] = ids_s[(rb + g) * 15 + off + j];
      v8f acc = {};
      for (int e = 0; e < 4; ++e) {
        if (e < 3) { // warm L2/L0 for next expert's weights
          __builtin_prefetch(W1t + ((e + 1) * 256 + wave * 32) * 256, 0, 1);
          __builtin_prefetch(W2t + ((e + 1) * 128 + wave * 16) * 256, 0, 1);
        }
        // GEMM1: (16x256)x(256x256); 16 col tiles over 8 waves
        #pragma unroll
        for (int ct = 0; ct < 2; ++ct) {
          const int c0 = (wave * 2 + ct) * 16;
          v8f c = {};
          #pragma unroll
          for (int k0 = 0; k0 < 256; k0 += 32) {
            TileBF a = load_tile(bin + j * 256 + k0, sin);
            TileBF w = load_tile(W1t + (e * 256 + c0) * 256 + k0, 256);
            c = wmma_bf16(a, w, c);
          }
          const float bias = b1[e * 256 + c0 + col];
          #pragma unroll
          for (int g = 0; g < 8; ++g)
            h1b[(rb + g) * SH1 + c0 + col] = f2us(fmaxf(c[g] + bias, 0.f));
        }
        __syncthreads();
        // GEMM2: (16x256)x(256x128); 8 col tiles over 8 waves, masked accumulate
        {
          const int c0 = wave * 16;
          v8f c = {};
          #pragma unroll
          for (int k0 = 0; k0 < 256; k0 += 32) {
            TileBF a = load_tile(h1b + k0, SH1);
            TileBF w = load_tile(W2t + (e * 128 + c0) * 256 + k0, 256);
            c = wmma_bf16(a, w, c);
          }
          const float bias = b2[e * 128 + c0 + col];
          #pragma unroll
          for (int g = 0; g < 8; ++g)
            acc[g] += (idv[g] == e) ? (c[g] + bias) : 0.f;
        }
        __syncthreads();
      }
      const int c0 = wave * 16;
      if (lvl < 3) {
        #pragma unroll
        for (int g = 0; g < 8; ++g)
          bout[(rb + g) * sout + j * 128 + c0 + col] = f2us(acc[g]);
      } else {
        #pragma unroll
        for (int g = 0; g < 8; ++g)
          roots[((bb * 16 + rb + g) * T_ + t) * EDIM + c0 + col] = acc[g];
      }
      __syncthreads();
    }
    off += n;
  }
}

// ---------------------------------------------------------------------------
// K2: leaf-function MLP on root pairs. grid = (64 batch-blocks, 9 pairs)
__global__ __launch_bounds__(256) void stmt_kernel(
    const float* __restrict__ roots, const int* __restrict__ lf_ids,
    const unsigned short* __restrict__ W1t, const float* __restrict__ b1,
    const unsigned short* __restrict__ W2t, const float* __restrict__ b2,
    float* __restrict__ stmt) {
  __shared__ unsigned short xb[16 * SX];
  __shared__ unsigned short h1b[16 * SH2];
  __shared__ int ids_s[16];

  const int bb   = blockIdx.x;
  const int p    = blockIdx.y;     // pair 0..8
  const int tid  = threadIdx.x;
  const int wave = tid >> 5;
  const int lane = tid & 31;
  const int col  = lane & 15;
  const int rb   = (lane & 16) >> 1;

  if (tid < 16) ids_s[tid] = lf_ids[(bb * 16 + tid) * 9 + p];
  { // x[r][:] = roots[b][16+2p] ++ roots[b][16+2p+1], f32->bf16
    int r = tid >> 4, seg = tid & 15;
    int b = bb * 16 + r;
    #pragma unroll
    for (int d = 0; d < 16; ++d) {
      int dim  = seg * 16 + d;
      int tree = 16 + 2 * p + (dim >> 7);
      xb[r * SX + dim] = f2us(roots[(b * T_ + tree) * EDIM + (dim & 127)]);
    }
  }
  __syncthreads();

  int idv[8];
  #pragma unroll
  for (int g = 0; g < 8; ++g) idv[g] = ids_s[rb + g];

  v8f acc = {};
  for (int e = 0; e < 3; ++e) {
    if (e < 2) {
      __builtin_prefetch(W1t + ((e + 1) * 512 + wave * 64) * 256, 0, 1);
      __builtin_prefetch(W2t + ((e + 1) * 128 + wave * 16) * 512, 0, 1);
    }
    // GEMM1: (16x256)x(256x512); 32 col tiles over 8 waves
    #pragma unroll
    for (int ct = 0; ct < 4; ++ct) {
      const int c0 = (wave * 4 + ct) * 16;
      v8f c = {};
      #pragma unroll
      for (int k0 = 0; k0 < 256; k0 += 32) {
        TileBF a = load_tile(xb + k0, SX);
        TileBF w = load_tile(W1t + (e * 512 + c0) * 256 + k0, 256);
        c = wmma_bf16(a, w, c);
      }
      const float bias = b1[e * 512 + c0 + col];
      #pragma unroll
      for (int g = 0; g < 8; ++g)
        h1b[(rb + g) * SH2 + c0 + col] = f2us(fmaxf(c[g] + bias, 0.f));
    }
    __syncthreads();
    // GEMM2: (16x512)x(512x128)
    {
      const int c0 = wave * 16;
      v8f c = {};
      #pragma unroll
      for (int k0 = 0; k0 < 512; k0 += 32) {
        TileBF a = load_tile(h1b + k0, SH2);
        TileBF w = load_tile(W2t + (e * 128 + c0) * 512 + k0, 512);
        c = wmma_bf16(a, w, c);
      }
      const float bias = b2[e * 128 + c0 + col];
      #pragma unroll
      for (int g = 0; g < 8; ++g)
        acc[g] += (idv[g] == e) ? (c[g] + bias) : 0.f;
    }
    __syncthreads();
  }
  const int c0 = wave * 16;
  #pragma unroll
  for (int g = 0; g < 8; ++g)
    stmt[((bb * 16 + rb + g) * 9 + p) * EDIM + c0 + col] = acc[g];
}

// ---------------------------------------------------------------------------
// K3: three small attentions + concat. grid = B, 128 threads.
__device__ void attend(const float* keys, int n, const float* q,
                       float* dst, float* sc) {
  const int tid = threadIdx.x;
  if (tid < n) {
    float s = 0.f;
    for (int d = 0; d < 128; ++d) s += keys[tid * 128 + d] * q[d];
    sc[tid] = s;
  }
  __syncthreads();
  float m = -3.0e38f;
  for (int i = 0; i < n; ++i) m = fmaxf(m, sc[i]);
  float sum = 0.f;
  for (int i = 0; i < n; ++i) sum += __expf(sc[i] - m);
  float ctx = 0.f;
  for (int i = 0; i < n; ++i) ctx += __expf(sc[i] - m) * keys[i * 128 + tid];
  dst[tid] = ctx / sum;
  __syncthreads();
}

__global__ __launch_bounds__(128) void attn_kernel(
    const float* __restrict__ roots, const float* __restrict__ stmt,
    const float* __restrict__ th_emb, const int* __restrict__ th_idx,
    const float* __restrict__ heW, const float* __restrict__ heB,
    const float* __restrict__ hgW, const float* __restrict__ hgB,
    const float* __restrict__ htW, const float* __restrict__ htB,
    float* __restrict__ out) {
  __shared__ float obj_s[128];
  __shared__ float q_s[3][128];
  __shared__ float thk[8 * 128];
  __shared__ float sc[16];

  const int b = blockIdx.x;
  const int tid = threadIdx.x;

  obj_s[tid] = stmt[(b * 9 + 8) * EDIM + tid];       // obj = stmt[:, GT_MAX]
  #pragma unroll
  for (int n2 = 0; n2 < 8; ++n2)
    thk[n2 * 128 + tid] = th_emb[th_idx[b * 8 + n2] * EDIM + tid];
  __syncthreads();

  float qe = heB[tid], qg = hgB[tid], qt = htB[tid];
  for (int i = 0; i < 128; ++i) {
    float o = obj_s[i];
    qe += o * heW[i * 128 + tid];
    qg += o * hgW[i * 128 + tid];
    qt += o * htW[i * 128 + tid];
  }
  q_s[0][tid] = qe; q_s[1][tid] = qg; q_s[2][tid] = qt;
  out[b * 512 + tid] = obj_s[tid];                   // [0:128] = obj
  __syncthreads();

  attend(stmt  + b * 9  * EDIM,  8, q_s[1], out + b * 512 + 128, sc); // gt_ctx
  attend(roots + b * T_ * EDIM, 16, q_s[0], out + b * 512 + 256, sc); // ent_ctx
  attend(thk,                    8, q_s[2], out + b * 512 + 384, sc); // th_ctx
}

// ---------------------------------------------------------------------------
extern "C" void kernel_launch(void* const* d_in, const int* in_sizes, int n_in,
                              void* d_out, int out_size, void* d_ws, size_t ws_size,
                              hipStream_t stream) {
  const int*   leaf_idx = (const int*)d_in[0];
  const int*   nf_ids   = (const int*)d_in[1];
  const int*   lf_ids   = (const int*)d_in[2];
  const int*   th_idx   = (const int*)d_in[3];
  const float* ent_emb  = (const float*)d_in[4];
  const float* th_emb   = (const float*)d_in[5];
  const float* nf_W1    = (const float*)d_in[6];
  const float* nf_b1    = (const float*)d_in[7];
  const float* nf_W2    = (const float*)d_in[8];
  const float* nf_b2    = (const float*)d_in[9];
  const float* lf_W1    = (const float*)d_in[10];
  const float* lf_b1    = (const float*)d_in[11];
  const float* lf_W2    = (const float*)d_in[12];
  const float* lf_b2    = (const float*)d_in[13];
  const float* heW = (const float*)d_in[14];
  const float* heB = (const float*)d_in[15];
  const float* hgW = (const float*)d_in[16];
  const float* hgB = (const float*)d_in[17];
  const float* htW = (const float*)d_in[18];
  const float* htB = (const float*)d_in[19];

  char* ws = (char*)d_ws;
  unsigned short* nfW1t = (unsigned short*)(ws + 0);        //  512 KB
  unsigned short* nfW2t = (unsigned short*)(ws + 524288);   //  256 KB
  unsigned short* lfW1t = (unsigned short*)(ws + 786432);   //  768 KB
  unsigned short* lfW2t = (unsigned short*)(ws + 1572864);  //  384 KB
  unsigned short* ent16 = (unsigned short*)(ws + 1966080);  //  512 KB (padded)
  float*          roots = (float*)(ws + 2490368);           // ~17.0 MB
  float*          stmt  = (float*)(ws + 20316160);          // ~4.5 MB

  // K0: weight convert + transpose to bf16 [e][out][in]; embedding table bf16
  wcvt_kernel<<<(4 * 256 * 256 + 255) / 256, 256, 0, stream>>>(
      nf_W1, nfW1t, 256 * 256, 256, 256, 4 * 256 * 256);
  wcvt_kernel<<<(4 * 256 * 128 + 255) / 256, 256, 0, stream>>>(
      nf_W2, nfW2t, 256 * 128, 256, 128, 4 * 256 * 128);
  wcvt_kernel<<<(3 * 256 * 512 + 255) / 256, 256, 0, stream>>>(
      lf_W1, lfW1t, 256 * 512, 256, 512, 3 * 256 * 512);
  wcvt_kernel<<<(3 * 512 * 128 + 255) / 256, 256, 0, stream>>>(
      lf_W2, lfW2t, 512 * 128, 512, 128, 3 * 512 * 128);
  ecvt_kernel<<<(2000 * 128 + 255) / 256, 256, 0, stream>>>(
      ent_emb, ent16, 2000 * 128);

  // K1: tree reduction -> roots
  tree_kernel<<<dim3(B_ / 16, T_), 256, 0, stream>>>(
      leaf_idx, nf_ids, ent_emb, ent16, nfW1t, nf_b1, nfW2t, nf_b2, roots);

  // K2: leaf-function MLP -> stmt
  stmt_kernel<<<dim3(B_ / 16, 9), 256, 0, stream>>>(
      roots, lf_ids, lfW1t, lf_b1, lfW2t, lf_b2, stmt);

  // K3: attentions + concat -> out (B, 512) f32
  attn_kernel<<<B_, 128, 0, stream>>>(
      roots, stmt, th_emb, th_idx, heW, heB, hgW, hgB, htW, htB, (float*)d_out);
}
